// ContrastLoss_Disentangle_54477365182707
// MI455X (gfx1250) — compile-verified, matrix operand order checked by reading, not run on hardware
//
#include <hip/hip_runtime.h>
#include <hip/hip_bf16.h>

typedef __attribute__((ext_vector_type(16))) _Float16 v16h;
typedef __attribute__((ext_vector_type(8)))  _Float16 half8;
typedef __attribute__((ext_vector_type(8)))  float    v8f;

// Problem constants
#define CC   8
#define NP   2048
#define NN   8192
#define DD   256
#define KK   4
#define KD   2048   // C*D, GEMM reduction length

// Workspace layout (bytes)
#define OFF_ACC 0                 // float[4]: normp_sum, normn_sum, bce_sum, lt_sum ; int at +16: notfind
#define OFF_X   256               // f16 X[2048][2048]
#define OFF_NPN (OFF_X   + (size_t)NP*KD*2)        // float norm_p[C][NP]
#define OFF_NNN (OFF_NPN + (size_t)CC*NP*4)        // float norm_n[C][NN]
#define OFF_PM  (OFF_NNN + (size_t)CC*NN*4)        // float pm[NP][NP]
#define OFF_SC  (OFF_PM  + (size_t)NP*NP*4)        // float scores[NN]
#define OFF_FUR (OFF_SC  + (size_t)NN*4)           // int furthest[NP]

__device__ __forceinline__ float wave_sum(float v) {
    #pragma unroll
    for (int off = 16; off >= 1; off >>= 1) v += __shfl_xor(v, off, 32);
    return v;
}

__global__ void init_kernel(float* acc, int* nf, float* out) {
    acc[0] = 0.f; acc[1] = 0.f; acc[2] = 0.f; acc[3] = 0.f;
    *nf = 0;
    out[0] = 0.f; out[1] = 0.f; out[2] = 0.f;
}

// One wave per (c,p): norm, atomic norm-sum, and write normalized f16 row chunk
// into X[p][c*256 + d].
__global__ __launch_bounds__(256) void pose_norm_kernel(const float* __restrict__ pose,
                                                        float* __restrict__ norm_p,
                                                        _Float16* __restrict__ X,
                                                        float* __restrict__ acc) {
    const int wave = threadIdx.x >> 5, lane = threadIdx.x & 31;
    const int idx = blockIdx.x * 8 + wave;          // 0 .. C*NP-1
    const int c = idx >> 11, p = idx & (NP - 1);
    const float* src = pose + ((size_t)(c * NP + p)) * DD + lane * 8;
    float4 v0 = *(const float4*)(src);
    float4 v1 = *(const float4*)(src + 4);
    float ss = v0.x*v0.x + v0.y*v0.y + v0.z*v0.z + v0.w*v0.w
             + v1.x*v1.x + v1.y*v1.y + v1.z*v1.z + v1.w*v1.w;
    ss = wave_sum(ss);
    const float nrm = sqrtf(ss);
    const float inv = 1.0f / nrm;
    if (lane == 0) { norm_p[c * NP + p] = nrm; atomicAdd(&acc[0], nrm); }
    _Float16* dst = X + (size_t)p * KD + c * DD + lane * 8;
    dst[0] = (_Float16)(v0.x * inv); dst[1] = (_Float16)(v0.y * inv);
    dst[2] = (_Float16)(v0.z * inv); dst[3] = (_Float16)(v0.w * inv);
    dst[4] = (_Float16)(v1.x * inv); dst[5] = (_Float16)(v1.y * inv);
    dst[6] = (_Float16)(v1.z * inv); dst[7] = (_Float16)(v1.w * inv);
}

// One wave per (c,n): norm of nlp row + atomic norm-sum.
__global__ __launch_bounds__(256) void nlp_norm_kernel(const float* __restrict__ nlp,
                                                       float* __restrict__ norm_n,
                                                       float* __restrict__ acc) {
    const int wave = threadIdx.x >> 5, lane = threadIdx.x & 31;
    const int idx = blockIdx.x * 8 + wave;          // 0 .. C*NN-1
    const int c = idx >> 13, n = idx & (NN - 1);
    const float* src = nlp + ((size_t)(c * NN + n)) * DD + lane * 8;
    float4 v0 = *(const float4*)(src);
    float4 v1 = *(const float4*)(src + 4);
    float ss = v0.x*v0.x + v0.y*v0.y + v0.z*v0.z + v0.w*v0.w
             + v1.x*v1.x + v1.y*v1.y + v1.z*v1.z + v1.w*v1.w;
    ss = wave_sum(ss);
    if (lane == 0) {
        const float nrm = sqrtf(ss);
        norm_n[c * NN + n] = nrm;
        atomicAdd(&acc[1], nrm);
    }
}

// pm = X * X^T, 128x128 tile per workgroup, 8 waves, each wave 32x64 (2x4 WMMA tiles).
#define LDSS 40   // LDS row stride in halves (80B, 16B-aligned rows)
__global__ __launch_bounds__(256) void pm_gemm_kernel(const _Float16* __restrict__ X,
                                                      float* __restrict__ pm) {
    __shared__ _Float16 As[128 * LDSS];
    __shared__ _Float16 Bs[128 * LDSS];
    const int t = threadIdx.x;
    const int lane = t & 31, wave = t >> 5;
    const int wm = wave >> 1;         // 0..3  -> 32-row strip
    const int wn = wave & 1;          // 0..1  -> 64-col strip
    const int rowBase = blockIdx.x * 128;
    const int colBase = blockIdx.y * 128;
    const int lrow = lane & 15;
    const int hs = lane >> 4;         // half-wave select
    const int kOff = hs * 8;          // lanes 16-31 read K 8..15 / 24..31

    v8f acc[2][4];
    #pragma unroll
    for (int mt = 0; mt < 2; ++mt)
        #pragma unroll
        for (int nt = 0; nt < 4; ++nt)
            #pragma unroll
            for (int i = 0; i < 8; ++i) acc[mt][nt][i] = 0.0f;

    for (int k0 = 0; k0 < KD; k0 += 32) {
        #pragma unroll
        for (int cchunk = t; cchunk < 512; cchunk += 256) {
            const int r = cchunk >> 2, kc = cchunk & 3;
            *(uint4*)(&As[r * LDSS + kc * 8]) =
                *(const uint4*)(X + (size_t)(rowBase + r) * KD + k0 + kc * 8);
            *(uint4*)(&Bs[r * LDSS + kc * 8]) =
                *(const uint4*)(X + (size_t)(colBase + r) * KD + k0 + kc * 8);
        }
        __syncthreads();

        v16h a[2], b[4];
        #pragma unroll
        for (int mt = 0; mt < 2; ++mt) {
            const int r = wm * 32 + mt * 16 + lrow;
            const half8 lo = *(const half8*)(&As[r * LDSS + kOff]);
            const half8 hi = *(const half8*)(&As[r * LDSS + kOff + 16]);
            #pragma unroll
            for (int i = 0; i < 8; ++i) { a[mt][i] = lo[i]; a[mt][8 + i] = hi[i]; }
        }
        #pragma unroll
        for (int nt = 0; nt < 4; ++nt) {
            const int r = wn * 64 + nt * 16 + lrow;
            const half8 lo = *(const half8*)(&Bs[r * LDSS + kOff]);
            const half8 hi = *(const half8*)(&Bs[r * LDSS + kOff + 16]);
            #pragma unroll
            for (int i = 0; i < 8; ++i) { b[nt][i] = lo[i]; b[nt][8 + i] = hi[i]; }
        }
        #pragma unroll
        for (int mt = 0; mt < 2; ++mt)
            #pragma unroll
            for (int nt = 0; nt < 4; ++nt)
                acc[mt][nt] = __builtin_amdgcn_wmma_f32_16x16x32_f16(
                    false, a[mt], false, b[nt], (short)0, acc[mt][nt], false, false);
        __syncthreads();
    }

    #pragma unroll
    for (int mt = 0; mt < 2; ++mt)
        #pragma unroll
        for (int nt = 0; nt < 4; ++nt)
            #pragma unroll
            for (int i = 0; i < 8; ++i) {
                const int row = rowBase + wm * 32 + mt * 16 + i + hs * 8;
                const int col = colBase + wn * 64 + nt * 16 + lrow;
                pm[(size_t)row * NP + col] = acc[mt][nt][i];
            }
}

// One wave per n: scores[n] = sum_c (nlpF . poseF[paired]) * categories[n][c]; BCE accum.
__global__ __launch_bounds__(256) void scores_kernel(const float* __restrict__ nlp,
                                                     const float* __restrict__ pose,
                                                     const float* __restrict__ norm_n,
                                                     const float* __restrict__ norm_p,
                                                     const int* __restrict__ nlpid2poseid,
                                                     const int* __restrict__ nlp_label,
                                                     const float* __restrict__ categories,
                                                     float* __restrict__ scores,
                                                     float* __restrict__ acc) {
    const int wave = threadIdx.x >> 5, lane = threadIdx.x & 31;
    const int n = blockIdx.x * 8 + wave;
    const int p = nlpid2poseid[n];
    float s = 0.f;
    #pragma unroll
    for (int c = 0; c < CC; ++c) {
        const float* an = nlp  + ((size_t)(c * NN + n)) * DD + lane * 8;
        const float* ap = pose + ((size_t)(c * NP + p)) * DD + lane * 8;
        float4 n0 = *(const float4*)(an),     n1 = *(const float4*)(an + 4);
        float4 p0 = *(const float4*)(ap),     p1 = *(const float4*)(ap + 4);
        float d = n0.x*p0.x + n0.y*p0.y + n0.z*p0.z + n0.w*p0.w
                + n1.x*p1.x + n1.y*p1.y + n1.z*p1.z + n1.w*p1.w;
        d = wave_sum(d);
        d /= (norm_n[c * NN + n] * norm_p[c * NP + p]);
        s += d * categories[(size_t)n * CC + c];
    }
    if (lane == 0) {
        scores[n] = s;
        const float y = (float)nlp_label[n];
        // loss = softplus(s) - s*y   (stable)
        const float sp = fmaxf(s, 0.f) + log1pf(expf(-fabsf(s)));
        atomicAdd(&acc[2], sp - s * y);
    }
}

// One workgroup per pose row: stable rank-r selection over pm row (diag forced to 1.0).
__global__ __launch_bounds__(256) void select_kernel(const float* __restrict__ pm,
                                                     const int* __restrict__ rand_index,
                                                     int* __restrict__ furthest) {
    __shared__ float row[NP];
    const int p = blockIdx.x, t = threadIdx.x;
    #pragma unroll
    for (int j = 0; j < 8; ++j) {
        const int q = t + j * 256;
        row[q] = (q == p) ? 1.0f : pm[(size_t)p * NP + q];
    }
    __syncthreads();
    const int r = rand_index[p];
    #pragma unroll
    for (int j = 0; j < 8; ++j) {
        const int q = t + j * 256;
        const float v = row[q];
        int cnt = 0;
        for (int q2 = 0; q2 < NP; ++q2) {
            const float w = row[q2];
            cnt += (w < v) || ((w == v) && (q2 < q));
        }
        if (cnt == r) furthest[p] = q;
    }
}

// One block (128 thr, 4 waves) per pose: cur[k] dots, then triple-loss logic.
__global__ __launch_bounds__(128) void triple_kernel(const float* __restrict__ nlp,
                                                     const float* __restrict__ pose,
                                                     const float* __restrict__ norm_n,
                                                     const float* __restrict__ norm_p,
                                                     const int* __restrict__ pose2nlpid,
                                                     const int* __restrict__ nlp_label,
                                                     const float* __restrict__ categories,
                                                     const float* __restrict__ scores,
                                                     const int* __restrict__ furthest,
                                                     float* __restrict__ acc,
                                                     int* __restrict__ notfind) {
    __shared__ float curv[KK];
    __shared__ int   lcurv[KK];
    const int p = blockIdx.x;
    const int k = threadIdx.x >> 5, lane = threadIdx.x & 31;
    const int fu = furthest[p];
    const int nid = pose2nlpid[fu * KK + k];
    float s = 0.f;
    #pragma unroll
    for (int c = 0; c < CC; ++c) {
        const float* an = nlp  + ((size_t)(c * NN + nid)) * DD + lane * 8;
        const float* ap = pose + ((size_t)(c * NP + p))   * DD + lane * 8;
        float4 n0 = *(const float4*)(an),  n1 = *(const float4*)(an + 4);
        float4 p0 = *(const float4*)(ap),  p1 = *(const float4*)(ap + 4);
        float d = n0.x*p0.x + n0.y*p0.y + n0.z*p0.z + n0.w*p0.w
                + n1.x*p1.x + n1.y*p1.y + n1.z*p1.z + n1.w*p1.w;
        d = wave_sum(d);
        d /= (norm_n[c * NN + nid] * norm_p[c * NP + p]);
        s += d * categories[(size_t)nid * CC + c];
    }
    if (lane == 0) { curv[k] = s; lcurv[k] = nlp_label[nid]; }
    __syncthreads();
    if (threadIdx.x == 0) {
        float mx = -INFINITY, mn = INFINITY;
        #pragma unroll
        for (int kk = 0; kk < KK; ++kk) {
            const int id = pose2nlpid[p * KK + kk];
            const float sg = scores[id];
            const int lg = nlp_label[id];
            if (lg == 0) mx = fmaxf(mx, sg);
            if (lg == 1) mn = fminf(mn, sg);
        }
        float maxp = fmaxf(mx, -1.0f);
        float minp = fminf(mn, 1.0f);
        float mc = -INFINITY;
        #pragma unroll
        for (int kk = 0; kk < KK; ++kk)
            if (lcurv[kk] == 1) mc = fmaxf(mc, curv[kk]);
        maxp = fmaxf(maxp, mc);
        const bool found = !((maxp == -1.0f) || (minp == 1.0f));
        if (found) atomicAdd(&acc[3], maxp - minp + 2.0f);
        else       atomicAdd(notfind, 1);
    }
}

__global__ void finalize_kernel(const float* acc, const int* notfind, float* out) {
    out[0] = acc[2] / (float)NN;                                  // loss_label
    out[1] = acc[0] / (float)(CC * NP) + acc[1] / (float)(CC * NN); // loss_norm
    const int nf = *notfind;
    out[2] = (nf == NN) ? 0.0f : acc[3] / (float)(NN - nf);       // loss_triple
}

extern "C" void kernel_launch(void* const* d_in, const int* in_sizes, int n_in,
                              void* d_out, int out_size, void* d_ws, size_t ws_size,
                              hipStream_t stream) {
    (void)in_sizes; (void)n_in; (void)out_size; (void)ws_size;
    const float* nlp          = (const float*)d_in[0];
    const float* pose         = (const float*)d_in[1];
    const int*   nlp_label    = (const int*)d_in[3];
    const int*   nlpid2poseid = (const int*)d_in[4];
    const int*   pose2nlpid   = (const int*)d_in[5];
    const float* categories   = (const float*)d_in[6];
    const int*   rand_index   = (const int*)d_in[7];
    float* out = (float*)d_out;

    char* ws = (char*)d_ws;
    float*     acc      = (float*)(ws + OFF_ACC);
    int*       notfind  = (int*)(ws + OFF_ACC + 16);
    _Float16*  X        = (_Float16*)(ws + OFF_X);
    float*     norm_p   = (float*)(ws + OFF_NPN);
    float*     norm_n   = (float*)(ws + OFF_NNN);
    float*     pm       = (float*)(ws + OFF_PM);
    float*     scores   = (float*)(ws + OFF_SC);
    int*       furthest = (int*)(ws + OFF_FUR);

    init_kernel<<<1, 1, 0, stream>>>(acc, notfind, out);
    pose_norm_kernel<<<CC * NP / 8, 256, 0, stream>>>(pose, norm_p, X, acc);
    nlp_norm_kernel <<<CC * NN / 8, 256, 0, stream>>>(nlp, norm_n, acc);
    pm_gemm_kernel  <<<dim3(NP / 128, NP / 128), 256, 0, stream>>>(X, pm);
    scores_kernel   <<<NN / 8, 256, 0, stream>>>(nlp, pose, norm_n, norm_p,
                                                 nlpid2poseid, nlp_label, categories,
                                                 scores, acc);
    select_kernel   <<<NP, 256, 0, stream>>>(pm, rand_index, furthest);
    triple_kernel   <<<NP, 128, 0, stream>>>(nlp, pose, norm_n, norm_p,
                                             pose2nlpid, nlp_label, categories,
                                             scores, furthest, acc, notfind);
    finalize_kernel <<<1, 1, 0, stream>>>(acc, notfind, out);
}
